// WindowAttention_Video_42580305773066
// MI455X (gfx1250) — matrix operand order repaired and checked by reading, not compiled, not run on hardware
//
#include <hip/hip_runtime.h>

typedef __bf16 bf16;
typedef bf16  v16bf_t __attribute__((ext_vector_type(16)));
typedef float v8f_t   __attribute__((ext_vector_type(8)));

__device__ __forceinline__ v8f_t zero8() {
  v8f_t z;
#pragma unroll
  for (int i = 0; i < 8; i++) z[i] = 0.f;
  return z;
}

// ---- CDNA5 async global->LDS copy (ASYNCcnt path) -------------------------
// dst lane LDS byte address = low 32 bits of generic LDS pointer.
#define ASYNC_LOAD_B128(ldsptr, gptr)                                        \
  asm volatile("global_load_async_to_lds_b128 %0, %1, off"                   \
               :: "v"((unsigned)(uintptr_t)(ldsptr)),                        \
                  "v"((unsigned long long)(uintptr_t)(gptr))                 \
               : "memory")
#define ASYNC_WAIT() asm volatile("s_wait_asynccnt 0" ::: "memory")

// A fragment (16x32 bf16, row-major in LDS, leading dim `ld`):
// lanes 0-15: M=lane, halves 0-7 = K 0..7, halves 8-15 = K 16..23
// lanes 16-31: M=lane-16, halves 0-7 = K 8..15, halves 8-15 = K 24..31
__device__ __forceinline__ v16bf_t load_a_frag(const bf16* base, int ld, int lane) {
  const int l = lane & 15, hi = lane >> 4;
  const bf16* r = base + l * ld + hi * 8;
  v16bf_t a;
#pragma unroll
  for (int i = 0; i < 8; i++) { a[i] = r[i]; a[8 + i] = r[16 + i]; }
  return a;
}

// B fragment (32x16 bf16, row-major in LDS): lane holds column N=lane%16,
// halves 0-7 = K 8h..8h+7, halves 8-15 = K 16+8h..16+8h+7 (h = lane/16)
__device__ __forceinline__ v16bf_t load_b_frag(const bf16* base, int ld, int lane) {
  const bf16* r = base + (lane >> 4) * 8 * ld + (lane & 15);
  v16bf_t b;
#pragma unroll
  for (int i = 0; i < 8; i++) { b[i] = r[i * ld]; b[8 + i] = r[(16 + i) * ld]; }
  return b;
}

#define WMMA_BF16(A, B, C) \
  __builtin_amdgcn_wmma_f32_16x16x32_bf16(false, (A), false, (B), (short)0, (C), false, false)

// ---------------------------------------------------------------------------
__global__ void cvt_f32_bf16_kernel(const float* __restrict__ s, bf16* __restrict__ d, int n) {
  int i = blockIdx.x * blockDim.x + threadIdx.x;
  if (i < n) d[i] = (bf16)s[i];
}

// ---------------------------------------------------------------------------
// GEMM: out = (X[rows x 512] @ W[512 x 256] + b) * scale, bf16 out.
// DUAL: second weight/out pair shares the X tile (fused K+V).
template <bool DUAL>
__global__ __launch_bounds__(256) void proj_kernel(
    const float* __restrict__ X, const bf16* __restrict__ W0,
    const float* __restrict__ b0, float scale0, bf16* __restrict__ out0,
    const bf16* __restrict__ W1, const float* __restrict__ b1, bf16* __restrict__ out1) {
  __shared__ bf16 Xs[64 * 32];
  __shared__ bf16 Ws0[32 * 256];
  __shared__ bf16 Ws1[32 * 256];
  const int tid = threadIdx.x, wave = tid >> 5, lane = tid & 31;
  const size_t m0 = (size_t)blockIdx.x * 64;

  v8f_t acc0[8], acc1[8];
#pragma unroll
  for (int i = 0; i < 8; i++) { acc0[i] = zero8(); acc1[i] = zero8(); }

  for (int kc = 0; kc < 512; kc += 32) {
#pragma unroll
    for (int j = 0; j < 4; j++) {  // W chunks 32x256 bf16 -> LDS (async copy)
      int v = tid + 256 * j, wr = v >> 5, g = v & 31;
      ASYNC_LOAD_B128(Ws0 + wr * 256 + g * 8, W0 + (size_t)(kc + wr) * 256 + g * 8);
      if (DUAL)
        ASYNC_LOAD_B128(Ws1 + wr * 256 + g * 8, W1 + (size_t)(kc + wr) * 256 + g * 8);
    }
    {  // X tile 64x32 fp32 -> bf16 LDS (needs conversion: VGPR path)
      int r = tid >> 2, c0 = (tid & 3) * 8;
      const float* xp = X + (m0 + r) * 512 + kc + c0;
      float4 f0 = *(const float4*)xp;
      float4 f1 = *(const float4*)(xp + 4);
      bf16* xd = Xs + r * 32 + c0;
      xd[0] = (bf16)f0.x; xd[1] = (bf16)f0.y; xd[2] = (bf16)f0.z; xd[3] = (bf16)f0.w;
      xd[4] = (bf16)f1.x; xd[5] = (bf16)f1.y; xd[6] = (bf16)f1.z; xd[7] = (bf16)f1.w;
    }
    ASYNC_WAIT();
    __syncthreads();
#pragma unroll
    for (int mt = 0; mt < 4; mt++) {
      v16bf_t a = load_a_frag(Xs + mt * 16 * 32, 32, lane);
#pragma unroll
      for (int j = 0; j < 2; j++) {
        int nt = wave + 8 * j;
        v16bf_t bb = load_b_frag(Ws0 + nt * 16, 256, lane);
        acc0[mt * 2 + j] = WMMA_BF16(a, bb, acc0[mt * 2 + j]);
        if (DUAL) {
          v16bf_t bb1 = load_b_frag(Ws1 + nt * 16, 256, lane);
          acc1[mt * 2 + j] = WMMA_BF16(a, bb1, acc1[mt * 2 + j]);
        }
      }
    }
    __syncthreads();
  }
  const int n = lane & 15, mb = (lane >> 4) * 8;
#pragma unroll
  for (int mt = 0; mt < 4; mt++)
#pragma unroll
    for (int j = 0; j < 2; j++) {
      int nt = wave + 8 * j, gn = nt * 16 + n;
      float bias0 = b0[gn];
      float bias1 = DUAL ? b1[gn] : 0.f;
#pragma unroll
      for (int vv = 0; vv < 8; vv++) {
        size_t gm = m0 + mt * 16 + mb + vv;
        out0[gm * 256 + gn] = (bf16)((acc0[mt * 2 + j][vv] + bias0) * scale0);
        if (DUAL) out1[gm * 256 + gn] = (bf16)(acc1[mt * 2 + j][vv] + bias1);
      }
    }
}

// ---------------------------------------------------------------------------
// One block per window t: S=Q·K^T+mask, softmax(147), O=P·V with d*49+s scatter.
__global__ __launch_bounds__(256) void attn_kernel(
    const bf16* __restrict__ Q, const bf16* __restrict__ K, const bf16* __restrict__ V,
    const float* __restrict__ mask, bf16* __restrict__ O) {
  extern __shared__ char smem[];
  bf16* Qs = (bf16*)smem;               // 64 x 256
  bf16* Kt = Qs + 64 * 256;             // 256 x 160 (d-major, keys padded)
  bf16* Vs = Kt + 256 * 160;            // 160 x 256 (rows 147..159 zero)
  float* Sf = (float*)(Vs + 160 * 256); // 64 x 160 (pre-staged with mask/-inf)
  bf16* Ps = (bf16*)(Sf + 64 * 160);    // 64 x 160

  const int tid = threadIdx.x, wave = tid >> 5, lane = tid & 31;
  const size_t t = blockIdx.x;
  const int widx = (int)(t & 63);
  const uint4 z4 = {0, 0, 0, 0};

  for (int v = tid; v < 64 * 32; v += 256) {  // Q: async copy, zero-pad rows 49..63
    int r = v >> 5, g = v & 31;
    if (r < 49) ASYNC_LOAD_B128(Qs + r * 256 + g * 8, Q + (t * 49 + r) * 256 + g * 8);
    else        *(uint4*)(Qs + r * 256 + g * 8) = z4;
  }
  for (int v = tid; v < 160 * 32; v += 256) {  // V: async copy, zero-pad rows 147..159
    int r = v >> 5, g = v & 31;
    if (r < 147) ASYNC_LOAD_B128(Vs + r * 256 + g * 8, V + (t * 147 + r) * 256 + g * 8);
    else         *(uint4*)(Vs + r * 256 + g * 8) = z4;
  }
  for (int v = tid; v < 147 * 32; v += 256) {  // K: transpose through VGPRs
    int j = v >> 5, d0 = (v & 31) * 8;
    union { uint4 u; bf16 h[8]; } tmp;
    tmp.u = *(const uint4*)(K + (t * 147 + j) * 256 + d0);
#pragma unroll
    for (int i = 0; i < 8; i++) Kt[(d0 + i) * 160 + j] = tmp.h[i];
  }
  // Pre-stage S with mask (valid region) / -1e30 (padding): makes the score
  // epilogue a branch-free LDS += and removes per-element global mask loads.
  for (int idx = tid; idx < 64 * 160; idx += 256) {
    int gm = idx / 160, gn = idx - gm * 160;
    float val = -1e30f;
    if (gm < 49 && gn < 147) {
      int km = gn; if (km >= 98) km -= 98; else if (km >= 49) km -= 49;  // tile x3
      val = mask[widx * 2401 + gm * 49 + km];
    }
    Sf[idx] = val;
  }
  ASYNC_WAIT();
  __syncthreads();

  // Phase 1: S += Q·K^T, 4x10 tiles of 16x16, K-dim 256 (8 wmma steps)
#pragma unroll
  for (int i = 0; i < 5; i++) {
    int tt = wave + 8 * i;
    int mt = tt / 10, nt = tt % 10;
    v8f_t acc = zero8();
#pragma unroll
    for (int kc = 0; kc < 8; kc++) {
      v16bf_t a = load_a_frag(Qs + mt * 16 * 256 + kc * 32, 256, lane);
      v16bf_t b = load_b_frag(Kt + kc * 32 * 160 + nt * 16, 160, lane);
      acc = WMMA_BF16(a, b, acc);
    }
    int n = lane & 15, mb = (lane >> 4) * 8;
    int off0 = (mt * 16 + mb) * 160 + nt * 16 + n;
#pragma unroll
    for (int vv = 0; vv < 8; vv++) Sf[off0 + vv * 160] += acc[vv];
  }
  __syncthreads();

  // Phase 2: row softmax over 147 keys, 4 lanes per row (groups stay in-wave)
  if (tid < 196) {
    int row = tid >> 2, sub = tid & 3;
    float* srow = Sf + row * 160;
    float mx = -1e30f;
    for (int j = sub; j < 147; j += 4) mx = fmaxf(mx, srow[j]);
    mx = fmaxf(mx, __shfl_xor(mx, 1, 32));
    mx = fmaxf(mx, __shfl_xor(mx, 2, 32));
    float s = 0.f;
    for (int j = sub; j < 147; j += 4) {
      float e = __expf(srow[j] - mx);
      srow[j] = e;
      s += e;
    }
    s += __shfl_xor(s, 1, 32);
    s += __shfl_xor(s, 2, 32);
    float inv = 1.f / s;
    for (int j = sub; j < 147; j += 4) Ps[row * 160 + j] = (bf16)(srow[j] * inv);
    for (int j = 147 + sub; j < 160; j += 4) Ps[row * 160 + j] = (bf16)0.f;
  } else {
    int t2 = tid - 196;  // 60 threads cover 15 pad rows x 4 lanes
    int row = 49 + (t2 >> 2), sub = t2 & 3;
    for (int j = sub; j < 160; j += 4) Ps[row * 160 + j] = (bf16)0.f;
  }
  __syncthreads();

  // Phase 3: O = P·V, 4x16 tiles, K-dim 160 (5 wmma steps); scatter d*49+s
#pragma unroll
  for (int i = 0; i < 8; i++) {
    int tt = wave + 8 * i;
    int mt = tt >> 4, nt = tt & 15;
    v8f_t acc = zero8();
#pragma unroll
    for (int kc = 0; kc < 5; kc++) {
      v16bf_t a = load_a_frag(Ps + mt * 16 * 160 + kc * 32, 160, lane);
      v16bf_t b = load_b_frag(Vs + kc * 32 * 256 + nt * 16, 256, lane);
      acc = WMMA_BF16(a, b, acc);
    }
    int n = lane & 15, mb = (lane >> 4) * 8;
    int gn = nt * 16 + n;
#pragma unroll
    for (int vv = 0; vv < 8; vv++) {
      int gm = mt * 16 + mb + vv;
      if (gm < 49) O[t * 12544 + (size_t)gn * 49 + gm] = (bf16)acc[vv];
    }
  }
}

// ---------------------------------------------------------------------------
// out = A[100352 x 256](bf16) @ Wp[256 x 256](bf16) + bp, fp32 out
__global__ __launch_bounds__(256) void out_proj_kernel(
    const bf16* __restrict__ A, const bf16* __restrict__ W,
    const float* __restrict__ bias, float* __restrict__ out) {
  __shared__ bf16 Xs[64 * 32];
  __shared__ bf16 Ws[32 * 256];
  const int tid = threadIdx.x, wave = tid >> 5, lane = tid & 31;
  const size_t m0 = (size_t)blockIdx.x * 64;
  v8f_t acc[8];
#pragma unroll
  for (int i = 0; i < 8; i++) acc[i] = zero8();

  for (int kc = 0; kc < 256; kc += 32) {
    {  // A tile: pure bf16 copy -> async path
      int r = tid >> 2, c0 = (tid & 3) * 8;
      ASYNC_LOAD_B128(Xs + r * 32 + c0, A + (m0 + r) * 256 + kc + c0);
    }
#pragma unroll
    for (int j = 0; j < 4; j++) {
      int v = tid + 256 * j, wr = v >> 5, g = v & 31;
      ASYNC_LOAD_B128(Ws + wr * 256 + g * 8, W + (size_t)(kc + wr) * 256 + g * 8);
    }
    ASYNC_WAIT();
    __syncthreads();
#pragma unroll
    for (int mt = 0; mt < 4; mt++) {
      v16bf_t a = load_a_frag(Xs + mt * 16 * 32, 32, lane);
#pragma unroll
      for (int j = 0; j < 2; j++) {
        int nt = wave + 8 * j;
        v16bf_t bb = load_b_frag(Ws + nt * 16, 256, lane);
        acc[mt * 2 + j] = WMMA_BF16(a, bb, acc[mt * 2 + j]);
      }
    }
    __syncthreads();
  }
  const int n = lane & 15, mb = (lane >> 4) * 8;
#pragma unroll
  for (int mt = 0; mt < 4; mt++)
#pragma unroll
    for (int j = 0; j < 2; j++) {
      int nt = wave + 8 * j, gn = nt * 16 + n;
      float b = bias[gn];
#pragma unroll
      for (int vv = 0; vv < 8; vv++) {
        size_t gm = m0 + mt * 16 + mb + vv;
        out[gm * 256 + gn] = acc[mt * 2 + j][vv] + b;
      }
    }
}

// ---------------------------------------------------------------------------
extern "C" void kernel_launch(void* const* d_in, const int* in_sizes, int n_in,
                              void* d_out, int out_size, void* d_ws, size_t ws_size,
                              hipStream_t stream) {
  (void)in_sizes; (void)n_in; (void)out_size; (void)ws_size;
  const float* x    = (const float*)d_in[0];
  const float* mask = (const float*)d_in[1];
  const float* Wq   = (const float*)d_in[2];
  const float* bq   = (const float*)d_in[3];
  const float* Wk   = (const float*)d_in[4];
  const float* bk   = (const float*)d_in[5];
  const float* Wv   = (const float*)d_in[6];
  const float* bv   = (const float*)d_in[7];
  const float* Wp   = (const float*)d_in[8];
  const float* bp   = (const float*)d_in[9];
  float* out = (float*)d_out;

  char* ws = (char*)d_ws;
  size_t off = 0;
  auto take = [&](size_t bytes) {
    char* p = ws + off;
    off += (bytes + 255) & ~(size_t)255;
    return p;
  };
  bf16* wq_b = (bf16*)take((size_t)512 * 256 * 2);
  bf16* wk_b = (bf16*)take((size_t)512 * 256 * 2);
  bf16* wv_b = (bf16*)take((size_t)512 * 256 * 2);
  bf16* wp_b = (bf16*)take((size_t)256 * 256 * 2);
  bf16* Qb   = (bf16*)take((size_t)100352 * 256 * 2);
  bf16* Kb   = (bf16*)take((size_t)301056 * 256 * 2);
  bf16* Vb   = (bf16*)take((size_t)301056 * 256 * 2);
  bf16* Ab   = (bf16*)take((size_t)100352 * 256 * 2);

  cvt_f32_bf16_kernel<<<512, 256, 0, stream>>>(Wq, wq_b, 512 * 256);
  cvt_f32_bf16_kernel<<<512, 256, 0, stream>>>(Wk, wk_b, 512 * 256);
  cvt_f32_bf16_kernel<<<512, 256, 0, stream>>>(Wv, wv_b, 512 * 256);
  cvt_f32_bf16_kernel<<<256, 256, 0, stream>>>(Wp, wp_b, 256 * 256);

  const float SCALE = 0.17677669529663687f;  // 32^-0.5

  // K and V projections fused (frames 0..2, rows = 301056)
  proj_kernel<true><<<4704, 256, 0, stream>>>(x, wk_b, bk, 1.0f, Kb, wv_b, bv, Vb);
  // Q projection (frame 3, rows = 100352), scale fused
  proj_kernel<false><<<1568, 256, 0, stream>>>(x + (size_t)3 * 51380224, wq_b, bq, SCALE, Qb,
                                               nullptr, nullptr, nullptr);

  // Attention: one block per window t, 252 KB dynamic LDS
  attn_kernel<<<2048, 256, 258048, stream>>>(Qb, Kb, Vb, mask, Ab);

  // Output projection -> fp32
  out_proj_kernel<<<1568, 256, 0, stream>>>(Ab, wp_b, bp, out);
}
